// SpikingCrossAttentionFormer_53480932769994
// MI455X (gfx1250) — compile-verified, matrix operand order checked by reading, not run on hardware
//
#include <hip/hip_runtime.h>

// ---------------------------------------------------------------------------
// SpikingCrossAttentionFormer on MI455X (gfx1250, wave32, WMMA + TDM).
// GEMMs use v_wmma_f32_16x16x32_f16 (activations are exact small integers ->
// f16 exact; weights rounded to f16 once per launch). Operand tiles are staged
// into LDS by the Tensor Data Mover (tensor_load_to_lds, TENSORcnt) when the
// builtin is available; otherwise by manual global->LDS staging.
// ---------------------------------------------------------------------------

typedef __attribute__((ext_vector_type(16))) _Float16 v16h;
typedef __attribute__((ext_vector_type(8)))  _Float16 v8h;
typedef __attribute__((ext_vector_type(8)))  float    v8f;
typedef __attribute__((ext_vector_type(4)))  unsigned u32x4;
typedef __attribute__((ext_vector_type(4)))  int      i32x4;
typedef __attribute__((ext_vector_type(8)))  int      i32x8;

#if defined(__gfx1250__) && __has_builtin(__builtin_amdgcn_tensor_load_to_lds)
  #define USE_TDM 1
  #if __has_include(<hip/amd_detail/amd_gfx1250_TDM.h>)
    #define TDM_ARGS6 1   // clang-23/therock header implies the 6-arg builtin
  #else
    #define TDM_ARGS6 0   // ROCm 7.2 clang-22: 5-arg builtin
  #endif
#else
  #define USE_TDM 0
#endif

namespace cfg {
constexpr int T_    = 4;
constexpr int Bb    = 32;
constexpr int Nn    = 256;
constexpr int Dd    = 128;
constexpr int HEADS = 2;
constexpr int DEPTH = 6;
constexpr int HID   = 512;           // 4*D
constexpr int Mrows = T_ * Bb * Nn;  // 32768 rows in the (T,B,N) "token" axis
constexpr int RowsT = Bb * Nn;       // 8192 rows per timestep
constexpr int BND   = Bb * Nn * Dd;  // 1048576
constexpr float EPS = 1e-5f;
constexpr int BLK = 128;             // GEMM M/N tile
constexpr int BK  = 64;              // GEMM K stage
constexpr int MT  = Mrows / BLK;     // 256 m-tiles
// transposed-f16 weight layout per (stream,layer)
constexpr int OFF_WQ = 0;
constexpr int OFF_WK = 16384;
constexpr int OFF_WP = 32768;
constexpr int OFF_WH = 49152;            // (1024 x 128): v rows 0..511, gate rows 512..1023
constexpr int OFF_WO = 49152 + 131072;   // (128 x 512)
constexpr int WLAYER = OFF_WO + 65536;   // 245760 halves
}
using namespace cfg;

// ---------------------------------------------------------------------------
// Weight prep: W (K x N, f32 row-major) -> Wt (N x K, f16 row-major)
// ---------------------------------------------------------------------------
__global__ void k_w_to_f16t(const float* __restrict__ W, _Float16* __restrict__ Wt,
                            int K, int N) {
  int id = blockIdx.x * 256 + threadIdx.x;
  if (id >= K * N) return;
  int k = id / N, n = id % N;
  Wt[(size_t)n * K + k] = (_Float16)W[id];
}

__global__ void k_to_f16(const float* __restrict__ X, _Float16* __restrict__ O, int n) {
  int id = blockIdx.x * 256 + threadIdx.x;
  if (id < n) O[id] = (_Float16)X[id];
}

// ---------------------------------------------------------------------------
// coder: layernorm over D (same for every t since input is broadcast), then
// LIF over T with scalar threshold. Writes binary spikes into residual stream.
// ---------------------------------------------------------------------------
__global__ __launch_bounds__(128) void k_coder(const float* __restrict__ inp,
                                               const float* __restrict__ g,
                                               const float* __restrict__ b,
                                               const float* __restrict__ thp,
                                               float* __restrict__ X) {
  int bn = blockIdx.x, d = threadIdx.x;
  __shared__ float sv[128], sq[128];
  float val = inp[(size_t)bn * Dd + d];
  sv[d] = val; sq[d] = val * val;
  __syncthreads();
  for (int s = 64; s > 0; s >>= 1) {
    if (d < s) { sv[d] += sv[d + s]; sq[d] += sq[d + s]; }
    __syncthreads();
  }
  float m   = sv[0] * (1.0f / Dd);
  float var = sq[0] * (1.0f / Dd) - m * m;
  float u   = (val - m) * rsqrtf(var + EPS) * g[d] + b[d];
  float th  = *thp;
  float v = 0.0f;
  for (int t = 0; t < T_; ++t) {
    v += (u - v) * 0.5f;
    float s = (v >= th) ? 1.0f : 0.0f;
    X[((size_t)(t * RowsT + bn)) * Dd + d] = s;
    v *= (1.0f - s);
  }
}

// ---------------------------------------------------------------------------
// Build a TDM D# (groups 0/1) describing one 2D f16 tile:
//   tile_dim0 = BK halves, tile_dim1 = BLK rows, row stride = K halves,
//   packed contiguously into LDS at lds_off.
// ---------------------------------------------------------------------------
#if USE_TDM
__device__ __forceinline__ u32x4 tdm_group0(unsigned lds_off, unsigned long long gaddr) {
  u32x4 g0;
  g0[0] = 1u;                                   // count=1 (valid user descriptor)
  g0[1] = lds_off;                              // lds_addr [63:32]
  g0[2] = (unsigned)gaddr;                      // global_addr [95:64]
  g0[3] = (unsigned)((gaddr >> 32) & 0x1FFFFFFu) | (2u << 30);  // addr hi + type=2
  return g0;
}
__device__ __forceinline__ i32x8 tdm_group1(unsigned K_) {
  const unsigned d0 = K_;            // tensor_dim0 (elements); tiles never OOB
  const unsigned d1 = 0x40000000u;   // tensor_dim1: large (no OOB on rows)
  i32x8 g1;
  g1[0] = (int)(1u << 16);                               // data_size=1 (2B), mask=0
  g1[1] = (int)((d0 & 0xFFFFu) << 16);                   // tensor_dim0[15:0]
  g1[2] = (int)((d0 >> 16) | ((d1 & 0xFFFFu) << 16));    // dim0 hi | dim1 lo
  g1[3] = (int)((d1 >> 16) | ((unsigned)BK << 16));      // dim1 hi | tile_dim0=64
  g1[4] = (int)(unsigned)BLK;                            // tile_dim1=128, tile_dim2=0
  g1[5] = (int)K_;                                       // tensor_dim0_stride lo
  g1[6] = 0;                                             // stride hi | dim1_stride lo
  g1[7] = 0;
  return g1;
}
__device__ __forceinline__ void tdm_load(u32x4 g0, i32x8 g1) {
  i32x4 z4 = {0, 0, 0, 0};
#if TDM_ARGS6
  i32x8 z8 = {0, 0, 0, 0, 0, 0, 0, 0};
  __builtin_amdgcn_tensor_load_to_lds(g0, g1, z4, z4, z8, 0);
#else
  __builtin_amdgcn_tensor_load_to_lds(g0, g1, z4, z4, 0);
#endif
}
#endif

// ---------------------------------------------------------------------------
// WMMA GEMM: C[M x Nout] = A[M x K](f16) * Bt[Nout x K](f16)^T + bias.
// Per workgroup: 128x128 tile, 8 waves (4M x 2N), per-wave 2x4 16x16 frags,
// K staged through LDS in 64-wide slices by the TDM (or manual fallback).
// Emits per-tile (sum, sum^2) channel partials via a deterministic
// lane->shfl->LDS->global reduction (no atomics).
// ---------------------------------------------------------------------------
template <int K, bool OUTH>
__global__ __launch_bounds__(256) void k_gemm(const _Float16* __restrict__ A,
                                              const _Float16* __restrict__ Bt,
                                              const float* __restrict__ bias,
                                              float* __restrict__ Cf,
                                              _Float16* __restrict__ Ch,
                                              float* __restrict__ pSum,
                                              float* __restrict__ pSq,
                                              int Nout) {
  __shared__ __align__(16) _Float16 sA[BLK][BK];
  __shared__ __align__(16) _Float16 sB[BLK][BK];
  __shared__ float sS[8][BLK];
  __shared__ float sQ[8][BLK];

  const int tid  = threadIdx.x;
  const int lane = tid & 31, wave = tid >> 5;
  const int wm = wave >> 1, wn = wave & 1;
  const int r = lane & 15, hs = lane >> 4;
  const int mBlk = blockIdx.x * BLK;
  const int nBlk = blockIdx.y * BLK;

  v8f acc[2][4] = {};

  const _Float16* gA = A  + (size_t)mBlk * K;
  const _Float16* gB = Bt + (size_t)nBlk * K;

#if USE_TDM
  const unsigned ldsA = (unsigned)(unsigned long long)(void*)&sA[0][0];
  const unsigned ldsB = (unsigned)(unsigned long long)(void*)&sB[0][0];
  const i32x8 g1 = tdm_group1((unsigned)K);
#endif

  for (int ks = 0; ks < K / BK; ++ks) {
#if USE_TDM
    if (wave == 0) {   // one wave drives the DMA; EXEC is ignored by TENSOR ops
      tdm_load(tdm_group0(ldsA, (unsigned long long)(const void*)(gA + ks * BK)), g1);
      tdm_load(tdm_group0(ldsB, (unsigned long long)(const void*)(gB + ks * BK)), g1);
      __builtin_amdgcn_s_wait_tensorcnt(0);
    }
#else
#pragma unroll
    for (int i = 0; i < 4; ++i) {
      int idx = tid + i * 256;          // 1024 16B vectors per operand tile
      int row = idx >> 3;
      int kv  = (idx & 7) * 8;
      *(v8h*)&sA[row][kv] = *(const v8h*)&gA[(size_t)row * K + ks * BK + kv];
      *(v8h*)&sB[row][kv] = *(const v8h*)&gB[(size_t)row * K + ks * BK + kv];
    }
#endif
    __syncthreads();
#pragma unroll
    for (int kk = 0; kk < BK / 32; ++kk) {
      const int k0 = kk * 32 + hs * 8;  // ISA 16-bit A layout: lanes<16 K0..7/16..23
      union { v16h v; v8h h[2]; } af[2], bf[4];
#pragma unroll
      for (int i = 0; i < 2; ++i) {
        int row = wm * 32 + i * 16 + r;
        af[i].h[0] = *(const v8h*)&sA[row][k0];
        af[i].h[1] = *(const v8h*)&sA[row][k0 + 16];
      }
#pragma unroll
      for (int j = 0; j < 4; ++j) {
        int row = wn * 64 + j * 16 + r;
        bf[j].h[0] = *(const v8h*)&sB[row][k0];
        bf[j].h[1] = *(const v8h*)&sB[row][k0 + 16];
      }
#pragma unroll
      for (int i = 0; i < 2; ++i)
#pragma unroll
        for (int j = 0; j < 4; ++j)
          acc[i][j] = __builtin_amdgcn_wmma_f32_16x16x32_f16(
              false, af[i].v, false, bf[j].v, (short)0, acc[i][j], false, false);
    }
    __syncthreads();
  }

  // Epilogue: bias add, store, and deterministic per-channel stats partials.
#pragma unroll
  for (int j = 0; j < 4; ++j) {
    const int nl = wn * 64 + j * 16 + r;   // each lane owns one channel per j
    const int gn = nBlk + nl;
    const float bv = bias[gn];
    float s = 0.0f, q = 0.0f;
#pragma unroll
    for (int i = 0; i < 2; ++i) {
      const int gm0 = mBlk + wm * 32 + i * 16 + hs * 8;  // C/D layout: vgpr e -> M=e(+8)
#pragma unroll
      for (int e = 0; e < 8; ++e) {
        float v = acc[i][j][e] + bv;
        if constexpr (OUTH) Ch[(size_t)(gm0 + e) * Nout + gn] = (_Float16)v;
        else                Cf[(size_t)(gm0 + e) * Nout + gn] = v;
        s += v; q += v * v;
      }
    }
    if constexpr (!OUTH) {
      s += __shfl_xor(s, 16);     // combine the two lanes sharing channel gn
      q += __shfl_xor(q, 16);
      if (hs == 0) { sS[wave][nl] = s; sQ[wave][nl] = q; }
    }
  }
  if constexpr (!OUTH) {
    __syncthreads();
    if (tid < BLK) {
      const int w0 = (tid >= 64) ? 1 : 0;   // which wave column covers this channel
      float s = 0.0f, q = 0.0f;
#pragma unroll
      for (int wmi = 0; wmi < 4; ++wmi) { s += sS[wmi * 2 + w0][tid]; q += sQ[wmi * 2 + w0][tid]; }
      pSum[(size_t)blockIdx.x * Nout + nBlk + tid] = s;
      pSq [(size_t)blockIdx.x * Nout + nBlk + tid] = q;
    }
  }
}

// fixed-order reduction of per-tile partials -> mean, inv-std per channel
__global__ void k_stats(const float* __restrict__ pS, const float* __restrict__ pQ,
                        float* __restrict__ mean, float* __restrict__ inv, int Nout) {
  int c = blockIdx.x * blockDim.x + threadIdx.x;
  if (c >= Nout) return;
  float s = 0.0f, q = 0.0f;
  for (int mt = 0; mt < MT; ++mt) { s += pS[(size_t)mt * Nout + c]; q += pQ[(size_t)mt * Nout + c]; }
  float m   = s * (1.0f / Mrows);
  float var = q * (1.0f / Mrows) - m * m;
  mean[c] = m;
  inv[c]  = rsqrtf(var + EPS);
}

// ---------------------------------------------------------------------------
// BN + LIF scan over T.  MODE 0: write binary spikes.  MODE 1: residual +=
// spikes.  MODE 2: gate(f16) *= spike (zero where spike==0).
// ---------------------------------------------------------------------------
template <int MODE>
__global__ void k_bn_lif(const float* __restrict__ Y, const float* __restrict__ mean,
                         const float* __restrict__ inv, const float* __restrict__ g,
                         const float* __restrict__ b, float th,
                         float* __restrict__ dst, _Float16* __restrict__ gate, int C) {
  int id = blockIdx.x * 256 + threadIdx.x;
  if (id >= RowsT * C) return;
  int c = id % C, bn = id / C;
  float m = mean[c], iv = inv[c] * g[c], bb = b[c];
  float v = 0.0f;
  for (int t = 0; t < T_; ++t) {
    size_t off = ((size_t)(t * RowsT + bn)) * C + c;
    float u = (Y[off] - m) * iv + bb;
    v += (u - v) * 0.5f;
    float s = (v >= th) ? 1.0f : 0.0f;
    if constexpr (MODE == 0)      dst[off] = s;
    else if constexpr (MODE == 1) dst[off] += s;
    else                          { if (s == 0.0f) gate[off] = (_Float16)0.0f; }
    v *= (1.0f - s);
  }
}

// a = lif(sum_q over the N/HEADS axis, th=0.5): one thread per (b,h,d)
__global__ void k_qsum_lif(const float* __restrict__ q, float* __restrict__ a) {
  int id = blockIdx.x * 256 + threadIdx.x;
  if (id >= Bb * HEADS * Dd) return;
  int d = id & 127, h = (id >> 7) & 1, b = id >> 8;
  float v = 0.0f;
  for (int t = 0; t < T_; ++t) {
    const float* qp = q + (((size_t)(t * Bb + b) * Nn + h * (Nn / HEADS)) * Dd) + d;
    float s = 0.0f;
    for (int nh = 0; nh < Nn / HEADS; ++nh) s += qp[(size_t)nh * Dd];
    v += (s - v) * 0.5f;
    float sp = (v >= 0.5f) ? 1.0f : 0.0f;
    a[((size_t)(t * Bb + b) * HEADS + h) * Dd + d] = sp;
    v *= (1.0f - sp);
  }
}

// out[t,b, nh*H + h, d] = a[t,b,h,d] * k[t,b, h*(N/H)+nh, d]  (binary product),
// written as f16 since it is the next GEMM's A operand.
__global__ void k_attn_mix(const float* __restrict__ a, const float* __restrict__ k,
                           _Float16* __restrict__ O) {
  int id = blockIdx.x * 256 + threadIdx.x;
  if (id >= Mrows * Dd) return;
  int d = id & 127;
  int m = id >> 7;
  int n = m % Nn, tb = m / Nn;
  int h = n & (HEADS - 1), nh = n >> 1;
  float av = a[((size_t)tb * HEADS + h) * Dd + d];
  float kv = k[(((size_t)tb * Nn) + h * (Nn / HEADS) + nh) * Dd + d];
  O[id] = (_Float16)(av * kv);
}

__global__ void k_final(const float* __restrict__ X, const float* __restrict__ Y,
                        const float* __restrict__ wi, const float* __restrict__ wt,
                        float* __restrict__ out) {
  int id = blockIdx.x * 256 + threadIdx.x;
  if (id >= BND) return;
  float oi = 0.0f, ot = 0.0f;
  for (int t = 0; t < T_; ++t) {
    oi += wi[t] * X[(size_t)t * BND + id];
    ot += wt[t] * Y[(size_t)t * BND + id];
  }
  out[id] = oi;
  out[BND + id] = ot;
}

// ---------------------------------------------------------------------------
extern "C" void kernel_launch(void* const* d_in, const int* in_sizes, int n_in,
                              void* d_out, int out_size, void* d_ws, size_t ws_size,
                              hipStream_t stream) {
  (void)in_sizes; (void)n_in; (void)out_size; (void)ws_size;
  const float* img = (const float*)d_in[0];
  const float* txt = (const float*)d_in[1];
  // params_i at d_in[2..21], params_t at d_in[22..41] in stack() insertion order:
  // 0 Wq,1 bq,2 q_bn_g,3 q_bn_b,4 Wk,5 bk,6 k_bn_g,7 k_bn_b,8 Wp,9 bp,
  // 10 p_bn_g,11 p_bn_b,12 Wh,13 bh,14 h_bn_g,15 h_bn_b,16 Wo,17 bo,18 o_bn_g,19 o_bn_b
  auto P = [&](int s, int idx) -> const float* { return (const float*)d_in[2 + s * 20 + idx]; };
  const float* ln_g1 = (const float*)d_in[42];
  const float* ln_b1 = (const float*)d_in[43];
  const float* th1p  = (const float*)d_in[44];
  const float* ln_g2 = (const float*)d_in[45];
  const float* ln_b2 = (const float*)d_in[46];
  const float* th2p  = (const float*)d_in[47];
  const float* w_img = (const float*)d_in[48];
  const float* w_txt = (const float*)d_in[49];

  // workspace carve (all buffers fully rewritten every call -> deterministic)
  char* w = (char*)d_ws;
  auto carve = [&](size_t bytes) { char* p = w; w += (bytes + 255) & ~(size_t)255; return p; };
  float*    xbuf  = (float*)carve((size_t)Mrows * Dd * 4);
  float*    ybuf  = (float*)carve((size_t)Mrows * Dd * 4);
  float*    Ybig  = (float*)carve((size_t)Mrows * HID * 4);
  float*    qspk  = (float*)carve((size_t)Mrows * Dd * 4);
  float*    kspk  = (float*)carve((size_t)Mrows * Dd * 4);
  float*    abuf  = (float*)carve((size_t)T_ * Bb * HEADS * Dd * 4);
  float*    pSum  = (float*)carve((size_t)MT * HID * 4);
  float*    pSq   = (float*)carve((size_t)MT * HID * 4);
  float*    meanA = (float*)carve(512 * 4);
  float*    invA  = (float*)carve(512 * 4);
  _Float16* A16   = (_Float16*)carve((size_t)Mrows * Dd * 2);
  _Float16* G16   = (_Float16*)carve((size_t)Mrows * HID * 2);
  _Float16* wf16  = (_Float16*)carve((size_t)2 * DEPTH * WLAYER * 2);

  // ---- weight prep: f32 -> f16 transposed ----
  for (int s = 0; s < 2; ++s)
    for (int l = 0; l < DEPTH; ++l) {
      _Float16* wb = wf16 + (size_t)(s * DEPTH + l) * WLAYER;
      k_w_to_f16t<<<64,  256, 0, stream>>>(P(s, 0)  + (size_t)l * Dd * Dd,      wb + OFF_WQ, Dd, Dd);
      k_w_to_f16t<<<64,  256, 0, stream>>>(P(s, 4)  + (size_t)l * Dd * Dd,      wb + OFF_WK, Dd, Dd);
      k_w_to_f16t<<<64,  256, 0, stream>>>(P(s, 8)  + (size_t)l * Dd * Dd,      wb + OFF_WP, Dd, Dd);
      k_w_to_f16t<<<512, 256, 0, stream>>>(P(s, 12) + (size_t)l * Dd * 2 * HID, wb + OFF_WH, Dd, 2 * HID);
      k_w_to_f16t<<<256, 256, 0, stream>>>(P(s, 16) + (size_t)l * HID * Dd,     wb + OFF_WO, HID, Dd);
    }

  // ---- coders ----
  k_coder<<<RowsT, 128, 0, stream>>>(img, ln_g1, ln_b1, th1p, xbuf);
  k_coder<<<RowsT, 128, 0, stream>>>(txt, ln_g2, ln_b2, th2p, ybuf);

  const int EL = Mrows * Dd;
  const int GB = (EL + 255) / 256;

  auto gemm128 = [&](const _Float16* Aop, const _Float16* Wt, const float* bias,
                     float* out, int Nout) {
    dim3 grid(MT, Nout / BLK);
    k_gemm<128, false><<<grid, 256, 0, stream>>>(Aop, Wt, bias, out, nullptr, pSum, pSq, Nout);
    k_stats<<<(Nout + 127) / 128, 128, 0, stream>>>(pSum, pSq, meanA, invA, Nout);
  };
  auto gemm128h = [&](const _Float16* Aop, const _Float16* Wt, const float* bias,
                      _Float16* out, int Nout) {
    dim3 grid(MT, Nout / BLK);
    k_gemm<128, true><<<grid, 256, 0, stream>>>(Aop, Wt, bias, nullptr, out, nullptr, nullptr, Nout);
  };
  auto gemm512 = [&](const _Float16* Aop, const _Float16* Wt, const float* bias,
                     float* out, int Nout) {
    dim3 grid(MT, Nout / BLK);
    k_gemm<512, false><<<grid, 256, 0, stream>>>(Aop, Wt, bias, out, nullptr, pSum, pSq, Nout);
    k_stats<<<(Nout + 127) / 128, 128, 0, stream>>>(pSum, pSq, meanA, invA, Nout);
  };
  auto to16 = [&](const float* src) { k_to_f16<<<GB, 256, 0, stream>>>(src, A16, EL); };

  auto attn_block = [&](float* X, const float* Ysrc, int s, int l) {
    _Float16* wb = wf16 + (size_t)(s * DEPTH + l) * WLAYER;
    to16(X);
    gemm128(A16, wb + OFF_WQ, P(s, 1) + l * Dd, Ybig, Dd);
    k_bn_lif<0><<<(RowsT * Dd + 255) / 256, 256, 0, stream>>>(
        Ybig, meanA, invA, P(s, 2) + l * Dd, P(s, 3) + l * Dd, 1.0f, qspk, nullptr, Dd);
    to16(Ysrc);
    gemm128(A16, wb + OFF_WK, P(s, 5) + l * Dd, Ybig, Dd);
    k_bn_lif<0><<<(RowsT * Dd + 255) / 256, 256, 0, stream>>>(
        Ybig, meanA, invA, P(s, 6) + l * Dd, P(s, 7) + l * Dd, 1.0f, kspk, nullptr, Dd);
    k_qsum_lif<<<(Bb * HEADS * Dd + 255) / 256, 256, 0, stream>>>(qspk, abuf);
    k_attn_mix<<<GB, 256, 0, stream>>>(abuf, kspk, A16);
    gemm128(A16, wb + OFF_WP, P(s, 9) + l * Dd, Ybig, Dd);
    k_bn_lif<1><<<(RowsT * Dd + 255) / 256, 256, 0, stream>>>(
        Ybig, meanA, invA, P(s, 10) + l * Dd, P(s, 11) + l * Dd, 1.0f, X, nullptr, Dd);
  };

  auto gfnn_block = [&](float* X, int s, int l) {
    _Float16* wb = wf16 + (size_t)(s * DEPTH + l) * WLAYER;
    to16(X);
    // v half (needs BN stats), f32 out
    gemm128(A16, wb + OFF_WH, P(s, 13) + l * 2 * HID, Ybig, HID);
    // gate half, straight to f16 (next GEMM's A operand)
    gemm128h(A16, wb + OFF_WH + (size_t)HID * Dd, P(s, 13) + l * 2 * HID + HID, G16, HID);
    // BN+LIF on v, multiply gate in place by binary spikes
    k_bn_lif<2><<<(RowsT * HID + 255) / 256, 256, 0, stream>>>(
        Ybig, meanA, invA, P(s, 14) + l * HID, P(s, 15) + l * HID, 1.0f, nullptr, G16, HID);
    gemm512(G16, wb + OFF_WO, P(s, 17) + l * Dd, Ybig, Dd);
    k_bn_lif<1><<<(RowsT * Dd + 255) / 256, 256, 0, stream>>>(
        Ybig, meanA, invA, P(s, 18) + l * Dd, P(s, 19) + l * Dd, 1.0f, X, nullptr, Dd);
  };

  for (int l = 0; l < DEPTH; ++l) {
    attn_block(xbuf, ybuf, 0, l);
    gfnn_block(xbuf, 0, l);
    attn_block(ybuf, xbuf, 1, l);
    gfnn_block(ybuf, 1, l);
  }

  k_final<<<(BND + 255) / 256, 256, 0, stream>>>(xbuf, ybuf, w_img, w_txt, (float*)d_out);
}
